// CSAAModule_76965813944626
// MI455X (gfx1250) — matrix operand order repaired and checked by reading, not compiled, not run on hardware
//
#include <hip/hip_runtime.h>
#include <hip/hip_bf16.h>

typedef __bf16 bf16;
typedef __attribute__((ext_vector_type(16))) __bf16 v16bf;
typedef __attribute__((ext_vector_type(8)))  float  v8f;

#define LDSPAD 48                      // row pitch (elements): 96 B, 16B-aligned
#define ABUF   (128 * LDSPAD)          // one A buffer, elements (128 rows x 32k)
#define BBUF   (128 * LDSPAD)          // one B buffer, elements (128 j x 32k)

__device__ __forceinline__ void async_b128(unsigned ldsOff, const void* g) {
    asm volatile("global_load_async_to_lds_b128 %0, %1, off"
                 :: "v"(ldsOff), "v"((unsigned long long)(uintptr_t)g)
                 : "memory");
}

// ---------------------------------------------------------------------------
// f32 -> bf16 conversion (grid-stride) for weights
// ---------------------------------------------------------------------------
__global__ __launch_bounds__(256) void csaa_cvt_bf16(const float* __restrict__ src,
                                                     bf16* __restrict__ dst,
                                                     long long n) {
    long long i = (long long)blockIdx.x * blockDim.x + threadIdx.x;
    long long stride = (long long)gridDim.x * blockDim.x;
    for (; i < n; i += stride) dst[i] = (bf16)src[i];
}

// ---------------------------------------------------------------------------
// Tiled transpose + convert: x[b][c][hw] (f32) -> xT[b][hw][c] (bf16)
// ---------------------------------------------------------------------------
__global__ __launch_bounds__(256) void csaa_cvt_tr(const float* __restrict__ src,
                                                   bf16* __restrict__ dst,
                                                   int C, int HW) {
    __shared__ float tile[32][33];
    const int b   = blockIdx.z;
    const int hw0 = blockIdx.x * 32;
    const int c0  = blockIdx.y * 32;
    const int tx  = threadIdx.x & 31;
    const int ty  = threadIdx.x >> 5;          // 0..7
    const float* s = src + (long long)b * C * HW;
    bf16* d = dst + (long long)b * C * HW;
    #pragma unroll
    for (int i = 0; i < 4; ++i)
        tile[ty + i * 8][tx] = s[(long long)(c0 + ty + i * 8) * HW + hw0 + tx];
    __syncthreads();
    #pragma unroll
    for (int i = 0; i < 4; ++i)
        d[(long long)(hw0 + ty + i * 8) * C + c0 + tx] = (bf16)tile[tx][ty + i * 8];
}

// ---------------------------------------------------------------------------
// Batched WMMA GEMM, K-innermost B operand, double-buffered async LDS staging.
//   A : bf16, addr = aBase + m*lda + k                    (K contiguous)
//   B : bf16, addr = bBase + nb*sBnHi + j*sBnLo + k       (K contiguous)
//        global n = nb*128 + j, nb = blockIdx.x (BN == 128)
//   D : f32/bf16, addr = dBase + m*sDm + nb*sDnHi + j*sDnLo
// Block: 256 thr = 8 wave32, 4(M) x 2(N); block tile 128(M) x 128(N); BK = 32.
// Wave tile 32x64 = 2x4 fragments -> 8 v_wmma_f32_16x16x32_bf16 per K-step
// from 12 ds_load_b128 (1.5 LDS reads per WMMA).
// ---------------------------------------------------------------------------
__global__ __launch_bounds__(256) void csaa_wmma_gemm(
    const bf16* __restrict__ A, long long aBO, long long aBI, int lda,
    const bf16* __restrict__ Bm, long long bBO, long long bBI,
    long long sBnHi, long long sBnLo,
    void* __restrict__ Dp, int dF32, long long dBO, long long dBI,
    long long sDm, long long sDnHi, long long sDnLo,
    const float* __restrict__ bias,
    int K, int batchI)
{
    __shared__ bf16 ldsA[2 * ABUF];    // [buf][row 0..127][k 0..31] pitch 48
    __shared__ bf16 ldsB[2 * BBUF];    // [buf][j   0..127][k 0..31] pitch 48

    const int bz = blockIdx.z;
    const int bo = bz / batchI;
    const int bi = bz - bo * batchI;

    const bf16* __restrict__ Ab = A  + (long long)bo * aBO + (long long)bi * aBI;
    const bf16* __restrict__ Bb = Bm + (long long)bo * bBO + (long long)bi * bBI;
    const long long dBase = (long long)bo * dBO + (long long)bi * dBI;
    const long long nbOff = (long long)blockIdx.x * sBnHi;

    const int t    = threadIdx.x;
    const int wave = t >> 5;
    const int lane = t & 31;
    const int l16  = lane & 15;
    const int half = lane >> 4;
    const int wM   = wave >> 1;            // 0..3
    const int wN   = wave & 1;             // 0..1

    // ---- cooperative async-copy assignments (b128 chunks, K-contiguous) ----
    // A tile: 128 rows x 32k (64B) = 512 chunks, 2/thread (rows r and r+64).
    const int aRow = t >> 2, aCk = (t & 3) * 8;
    const unsigned lA0a = (unsigned)(uintptr_t)&ldsA[aRow * LDSPAD + aCk];
    const unsigned lA0b = lA0a + 64 * LDSPAD * 2;
    const unsigned lA1a = lA0a + ABUF * 2;
    const unsigned lA1b = lA0b + ABUF * 2;
    const bf16* gA0 = Ab + (long long)(blockIdx.y * 128 + aRow) * lda + aCk;
    const bf16* gA1 = Ab + (long long)(blockIdx.y * 128 + 64 + aRow) * lda + aCk;
    // B tile: 128 j-rows x 32k (64B) = 512 chunks, 2/thread (j and j+64).
    const unsigned lB0a = (unsigned)(uintptr_t)&ldsB[aRow * LDSPAD + aCk];
    const unsigned lB0b = lB0a + 64 * LDSPAD * 2;
    const unsigned lB1a = lB0a + BBUF * 2;
    const unsigned lB1b = lB0b + BBUF * 2;
    const bf16* gB0 = Bb + nbOff + (long long)aRow * sBnLo + aCk;
    const bf16* gB1 = Bb + nbOff + (long long)(aRow + 64) * sBnLo + aCk;

    v8f acc[2][4];
    #pragma unroll
    for (int fm = 0; fm < 2; ++fm)
        #pragma unroll
        for (int fn = 0; fn < 4; ++fn)
            acc[fm][fn] = (v8f){0.f,0.f,0.f,0.f,0.f,0.f,0.f,0.f};

    const int nSlab = K >> 5;

    // prologue: slab 0 -> buffer 0
    async_b128(lA0a, gA0);
    async_b128(lA0b, gA1);
    async_b128(lB0a, gB0);
    async_b128(lB0b, gB1);

    for (int s = 0; s < nSlab; ++s) {
        const int cur = s & 1;
        if (s + 1 < nSlab) {
            const long long kn = (long long)(s + 1) * 32;
            if (cur == 0) {
                async_b128(lA1a, gA0 + kn); async_b128(lA1b, gA1 + kn);
                async_b128(lB1a, gB0 + kn); async_b128(lB1b, gB1 + kn);
            } else {
                async_b128(lA0a, gA0 + kn); async_b128(lA0b, gA1 + kn);
                async_b128(lB0a, gB0 + kn); async_b128(lB0b, gB1 + kn);
            }
            asm volatile("s_wait_asynccnt 0x4" ::: "memory");  // current slab done
        } else {
            asm volatile("s_wait_asynccnt 0x0" ::: "memory");
        }
        __syncthreads();   // all waves' slab data visible

        const bf16* __restrict__ bufA = &ldsA[cur * ABUF];
        const bf16* __restrict__ bufB = &ldsB[cur * BBUF];

        // A fragments (16x32): k = i + (i&8) + 8*half -> two 16B reads/frag
        v16bf afr[2];
        #pragma unroll
        for (int fm = 0; fm < 2; ++fm) {
            const int r = (wM * 32 + fm * 16 + l16) * LDSPAD;
            #pragma unroll
            for (int i = 0; i < 8; ++i) {
                afr[fm][i]     = bufA[r + 8 * half + i];
                afr[fm][i + 8] = bufA[r + 16 + 8 * half + i];
            }
        }
        // B fragments (32x16): k = i + (i&8) + 8*half at column j -> two 16B reads/frag
        v16bf bfr[4];
        #pragma unroll
        for (int fn = 0; fn < 4; ++fn) {
            const int c = (wN * 64 + fn * 16 + l16) * LDSPAD + 8 * half;
            #pragma unroll
            for (int i = 0; i < 8; ++i) {
                bfr[fn][i]     = bufB[c + i];
                bfr[fn][i + 8] = bufB[c + 16 + i];
            }
        }

        #pragma unroll
        for (int fm = 0; fm < 2; ++fm)
            #pragma unroll
            for (int fn = 0; fn < 4; ++fn)
                acc[fm][fn] = __builtin_amdgcn_wmma_f32_16x16x32_bf16(
                    false, afr[fm], false, bfr[fn], (short)0, acc[fm][fn], false, false);

        __syncthreads();   // all waves done reading before buffer reuse
    }

    // ---- store: D layout m = d + 8*half, n = l16 per fragment ----
    const long long dHi = (long long)blockIdx.x * sDnHi;
    #pragma unroll
    for (int fm = 0; fm < 2; ++fm) {
        #pragma unroll
        for (int fn = 0; fn < 4; ++fn) {
            const int jn = wN * 64 + fn * 16 + l16;
            #pragma unroll
            for (int d = 0; d < 8; ++d) {
                const int mm = blockIdx.y * 128 + wM * 32 + fm * 16 + d + 8 * half;
                float r = acc[fm][fn][d];
                if (bias) r += bias[mm];
                const long long addr = dBase + (long long)mm * sDm + dHi + (long long)jn * sDnLo;
                if (dF32) ((float*)Dp)[addr] = r;
                else      ((bf16*)Dp)[addr] = (bf16)r;
            }
        }
    }
}

// ---------------------------------------------------------------------------
// Row softmax over 128 elements: one wave32 per row, bf16 probs out.
// ---------------------------------------------------------------------------
__global__ __launch_bounds__(128) void csaa_softmax128(const float* __restrict__ S,
                                                       bf16* __restrict__ P,
                                                       int rows) {
    const int row  = blockIdx.x * 4 + (threadIdx.x >> 5);
    const int lane = threadIdx.x & 31;
    if (row >= rows) return;

    const float* __restrict__ s = S + (long long)row * 128;
    float x0 = s[lane], x1 = s[lane + 32], x2 = s[lane + 64], x3 = s[lane + 96];

    float mx = fmaxf(fmaxf(x0, x1), fmaxf(x2, x3));
    #pragma unroll
    for (int off = 16; off; off >>= 1) mx = fmaxf(mx, __shfl_xor(mx, off, 32));

    x0 = __expf(x0 - mx); x1 = __expf(x1 - mx);
    x2 = __expf(x2 - mx); x3 = __expf(x3 - mx);
    float sum = x0 + x1 + x2 + x3;
    #pragma unroll
    for (int off = 16; off; off >>= 1) sum += __shfl_xor(sum, off, 32);
    const float inv = __frcp_rn(sum);

    bf16* __restrict__ p = P + (long long)row * 128;
    p[lane]      = (bf16)(x0 * inv);
    p[lane + 32] = (bf16)(x1 * inv);
    p[lane + 64] = (bf16)(x2 * inv);
    p[lane + 96] = (bf16)(x3 * inv);
}

// ---------------------------------------------------------------------------
// Host-side orchestration (all B operands K-innermost)
// ---------------------------------------------------------------------------
extern "C" void kernel_launch(void* const* d_in, const int* in_sizes, int n_in,
                              void* d_out, int out_size, void* d_ws, size_t ws_size,
                              hipStream_t stream) {
    const long long Bn = 8, CIN = 256, COUT = 256, R = 128;
    const long long HW = R * R;          // 16384
    const long long RHW = R * HW;        // 2,097,152
    const long long CHW = CIN * HW;      // 4,194,304

    const float* x   = (const float*)d_in[0];
    const float* Wr  = (const float*)d_in[1];
    const float* br  = (const float*)d_in[2];
    const float* Wqw = (const float*)d_in[3];
    const float* bqw = (const float*)d_in[4];
    const float* Wkw = (const float*)d_in[5];
    const float* bkw = (const float*)d_in[6];
    const float* Wvw = (const float*)d_in[7];
    const float* bvw = (const float*)d_in[8];
    const float* Wqh = (const float*)d_in[9];
    const float* bqh = (const float*)d_in[10];
    const float* Wkh = (const float*)d_in[11];
    const float* bkh = (const float*)d_in[12];
    const float* Wvh = (const float*)d_in[13];
    const float* bvh = (const float*)d_in[14];
    const float* Wo  = (const float*)d_in[15];
    const float* bo  = (const float*)d_in[16];

    char* ws = (char*)d_ws;
    bf16* xT  = (bf16*)(ws + 0);             // [B][HW][CIN]  (K-innermost)
    bf16* wrb = (bf16*)(ws + 67108864LL);
    bf16* wqwb= (bf16*)(ws + 67174400LL);
    bf16* wkwb= (bf16*)(ws + 67207168LL);
    bf16* wvwb= (bf16*)(ws + 67239936LL);
    bf16* wqhb= (bf16*)(ws + 67272704LL);
    bf16* wkhb= (bf16*)(ws + 67305472LL);
    bf16* wvhb= (bf16*)(ws + 67338240LL);
    bf16* wob = (bf16*)(ws + 67371008LL);
    bf16* y1  = (bf16*)(ws + 67436544LL);    // [B][r][w][h]; reused as y2 [B][h][w][c]
    bf16* qb  = (bf16*)(ws + 100990976LL);   // [B][o][r][w] natural
    bf16* kb  = (bf16*)(ws + 134545408LL);   // [B][o][u][w] natural
    bf16* vb  = (bf16*)(ws + 168099840LL);   // [B][o][w][u] TRANSPOSED
    float* Sb = (float*)(ws + 201654272LL);  // [B][o][r][u] f32 scores
    bf16* Pb  = (bf16*)(ws + 268763136LL);   // [B][o][r][u] bf16 probs
    bf16* ow  = (bf16*)(ws + 302317568LL);   // [B][r][h][w]

    // ---- 0) convert + transpose inputs ----
    csaa_cvt_tr<<<dim3(512, 8, 8), 256, 0, stream>>>(x, xT, 256, 16384);
    csaa_cvt_bf16<<<64, 256, 0, stream>>>(Wr,  wrb,  R * CIN);
    csaa_cvt_bf16<<<64, 256, 0, stream>>>(Wqw, wqwb, R * R);
    csaa_cvt_bf16<<<64, 256, 0, stream>>>(Wkw, wkwb, R * R);
    csaa_cvt_bf16<<<64, 256, 0, stream>>>(Wvw, wvwb, R * R);
    csaa_cvt_bf16<<<64, 256, 0, stream>>>(Wqh, wqhb, R * R);
    csaa_cvt_bf16<<<64, 256, 0, stream>>>(Wkh, wkhb, R * R);
    csaa_cvt_bf16<<<64, 256, 0, stream>>>(Wvh, wvhb, R * R);
    csaa_cvt_bf16<<<64, 256, 0, stream>>>(Wo,  wob,  COUT * R);

    // ---- 1) resize: y1[b][r][w][h] = Wr x xT + br ----
    csaa_wmma_gemm<<<dim3(128, 1, 8), 256, 0, stream>>>(
        wrb, 0, 0, 256,
        xT, CHW, 0, /*sBnHi=*/32768, /*sBnLo=*/256,
        y1, 0, RHW, 0, /*sDm=*/16384, /*sDnHi=*/1, /*sDnLo=*/128,
        br, 256, 1);

    // ---- 2) width projections from y1 (B: k=h innermost) ----
    csaa_wmma_gemm<<<dim3(128, 1, 8), 256, 0, stream>>>(      // q natural
        wqwb, 0, 0, 128, y1, RHW, 0, 16384, 128,
        qb, 0, RHW, 0, 16384, 128, 1, bqw, 128, 1);
    csaa_wmma_gemm<<<dim3(128, 1, 8), 256, 0, stream>>>(      // k natural
        wkwb, 0, 0, 128, y1, RHW, 0, 16384, 128,
        kb, 0, RHW, 0, 16384, 128, 1, bkw, 128, 1);
    csaa_wmma_gemm<<<dim3(128, 1, 8), 256, 0, stream>>>(      // v transposed
        wvwb, 0, 0, 128, y1, RHW, 0, 16384, 128,
        vb, 0, RHW, 0, 16384, /*hi=*/1, /*lo=*/128, bvw, 128, 1);

    // ---- 3) width scores: S = q . k^T (contraction w innermost both sides) ----
    csaa_wmma_gemm<<<dim3(1, 1, 1024), 256, 0, stream>>>(
        qb, RHW, 16384, 128,
        kb, RHW, 16384, /*sBnHi=*/0, /*sBnLo=*/128,
        Sb, 1, RHW, 16384, /*sDm=*/128, 0, 1,
        nullptr, 128, 128);
    csaa_softmax128<<<32768, 128, 0, stream>>>(Sb, Pb, 131072);

    // ---- 4) width attn-out: ow[b][r][h=o][w] = P x v^T ----
    csaa_wmma_gemm<<<dim3(1, 1, 1024), 256, 0, stream>>>(
        Pb, RHW, 16384, 128,
        vb, RHW, 16384, 0, 128,
        ow, 0, RHW, /*dBI=*/128, /*sDm=*/16384, 0, /*sDnLo=*/1,
        nullptr, 128, 128);

    // ---- 5) height projections from ow (B: k=w innermost) ----
    csaa_wmma_gemm<<<dim3(128, 1, 8), 256, 0, stream>>>(      // qh natural
        wqhb, 0, 0, 128, ow, RHW, 0, 16384, 128,
        qb, 0, RHW, 0, 16384, 128, 1, bqh, 128, 1);
    csaa_wmma_gemm<<<dim3(128, 1, 8), 256, 0, stream>>>(      // kh natural
        wkhb, 0, 0, 128, ow, RHW, 0, 16384, 128,
        kb, 0, RHW, 0, 16384, 128, 1, bkh, 128, 1);
    csaa_wmma_gemm<<<dim3(128, 1, 8), 256, 0, stream>>>(      // vh transposed
        wvhb, 0, 0, 128, ow, RHW, 0, 16384, 128,
        vb, 0, RHW, 0, 16384, 1, 128, bvh, 128, 1);

    // ---- 6) height scores + softmax ----
    csaa_wmma_gemm<<<dim3(1, 1, 1024), 256, 0, stream>>>(
        qb, RHW, 16384, 128,
        kb, RHW, 16384, 0, 128,
        Sb, 1, RHW, 16384, 128, 0, 1,
        nullptr, 128, 128);
    csaa_softmax128<<<32768, 128, 0, stream>>>(Sb, Pb, 131072);

    // ---- 7) height attn-out: y2[b][h][w=o2][c=m] (reuse y1 buffer) ----
    csaa_wmma_gemm<<<dim3(1, 1, 1024), 256, 0, stream>>>(
        Pb, RHW, 16384, 128,
        vb, RHW, 16384, 0, 128,
        y1, 0, RHW, /*dBI=*/128, /*sDm=*/1, 0, /*sDnLo=*/16384,
        nullptr, 128, 128);

    // ---- 8) restore: d_out[b][co][h][w] = Wo x y2 + bo (f32) ----
    csaa_wmma_gemm<<<dim3(128, 2, 8), 256, 0, stream>>>(
        wob, 0, 0, 128,
        y1, RHW, 0, 16384, 128,
        d_out, 1, COUT * HW, 0, 16384, 128, 1,
        bo, 128, 1);
}
